// VisualNodeEdgeMLPEnding_31533649887474
// MI455X (gfx1250) — compile-verified, matrix operand order checked by reading, not run on hardware
//
#include <hip/hip_runtime.h>
#include <hip/hip_bf16.h>
#include <stdint.h>

// ============================================================================
// VisualNodeEdgeMLPEnding for MI455X (gfx1250, wave32, WMMA).
// bf16 activations everywhere (op is HBM-bandwidth bound; 23.3 TB/s),
// NHWC CNN so pointwise convs + MLPs all hit one tiled WMMA bf16 GEMM.
// This round: global->LDS staging uses GLOBAL_LOAD_ASYNC_TO_LDS_B128
// (ASYNCcnt-tracked async DMA, no VGPR staging) + s_wait_asynccnt.
// ============================================================================

typedef __bf16 bf16;
typedef __attribute__((ext_vector_type(16))) __bf16 v16bf;
typedef __attribute__((ext_vector_type(8)))  float  v8f;

#define GN 4000
#define GE 16000

static inline int cdiv_i(int a, int b) { return (a + b - 1) / b; }

// async 16B global->LDS copy, GV mode; second instruction covers bytes 16..31.
// ISA: copies global bytes at VADDR+off into LDS at VDST+off; tracked by ASYNCcnt.
__device__ __forceinline__ void async_g2l_16(const bf16* g, bf16* l) {
  uint32_t lo = (uint32_t)(uintptr_t)l;          // low 32 bits of generic = LDS addr
  uint64_t ga = (uint64_t)(uintptr_t)g;
  asm volatile("global_load_async_to_lds_b128 %0, %1, off"
               :: "v"(lo), "v"(ga) : "memory");
}
__device__ __forceinline__ void async_g2l_32(const bf16* g, bf16* l) {
  uint32_t lo = (uint32_t)(uintptr_t)l;
  uint64_t ga = (uint64_t)(uintptr_t)g;
  asm volatile("global_load_async_to_lds_b128 %0, %1, off\n\t"
               "global_load_async_to_lds_b128 %0, %1, off offset:16"
               :: "v"(lo), "v"(ga) : "memory");
}
__device__ __forceinline__ void wait_async0() {
  asm volatile("s_wait_asynccnt 0x0" ::: "memory");
}

// ---------------------------------------------------------------------------
// Tiled GEMM: C[M,N] = A[M,K](bf16,row-major) * B[N,K](bf16,row-major)^T + bias
// BM=128, BN=64, BK=32; 256 threads = 8 waves (4x2); wave does 2x2 WMMA tiles.
// Double-buffered LDS; producer side is async-DMA, consumer is ds_load+WMMA.
// ---------------------------------------------------------------------------
template <int RELU, typename OutT>
__global__ __launch_bounds__(256) void k_gemm(
    const bf16* __restrict__ A, int lda,
    const bf16* __restrict__ Bm, int ldb,
    const float* __restrict__ bias,
    OutT* __restrict__ C, int ldc,
    int M, int N, int K)
{
  constexpr int BM = 128, BN = 64, BK = 32, PAD = 8; // 40-elem rows: conflict-free frags
  __shared__ bf16 As[2][BM][BK + PAD];
  __shared__ bf16 Bs[2][BN][BK + PAD];

  const int tid  = threadIdx.x;
  const int lane = tid & 31;
  const int wid  = tid >> 5;
  const int wm   = (wid >> 1) * 32;   // 0,32,64,96
  const int wn   = (wid & 1) * 32;    // 0,32
  const int blockM = blockIdx.y * BM;
  const int blockN = blockIdx.x * BN;

  v8f acc[2][2];
  for (int i = 0; i < 2; ++i)
    for (int j = 0; j < 2; ++j)
      for (int r = 0; r < 8; ++r) acc[i][j][r] = 0.f;

  // global->LDS staging assignments
  const int arow = tid >> 1;         // 0..127
  const int acol = (tid & 1) * 16;   // 0,16
  const int brow = tid >> 2;         // 0..63
  const int bcol = (tid & 3) * 8;    // 0,8,16,24

  auto loadTile = [&](int buf, int k0) {
    { // A tile: 32B per thread
      int gm = blockM + arow;
      const bf16* src = A + (long)gm * lda + k0 + acol;
      bf16* dst = &As[buf][arow][acol];
      bool fast = (gm < M) && (k0 + acol + 16 <= K) && ((((uintptr_t)src) & 15) == 0);
      if (fast) {
        async_g2l_32(src, dst);                      // ASYNCcnt path
      } else {
        for (int i = 0; i < 16; ++i)
          dst[i] = (gm < M && (k0 + acol + i) < K) ? src[i] : (bf16)0.f;
      }
    }
    { // B tile: 16B per thread
      int gn = blockN + brow;
      const bf16* src = Bm + (long)gn * ldb + k0 + bcol;
      bf16* dst = &Bs[buf][brow][bcol];
      bool fast = (gn < N) && (k0 + bcol + 8 <= K) && ((((uintptr_t)src) & 15) == 0);
      if (fast) {
        async_g2l_16(src, dst);
      } else {
        for (int i = 0; i < 8; ++i)
          dst[i] = (gn < N && (k0 + bcol + i) < K) ? src[i] : (bf16)0.f;
      }
    }
  };

  union Frag { v16bf v; uint4 q[2]; };
  const int khalf = lane >> 4;     // ISA 16-bit A/B layout: lanes 16-31 hold K+8
  const int rA    = lane & 15;

  const int nk = (K + BK - 1) / BK;
  loadTile(0, 0);
  wait_async0();
  __syncthreads();

  for (int kt = 0; kt < nk; ++kt) {
    int buf = kt & 1;
    if (kt + 1 < nk) loadTile(buf ^ 1, (kt + 1) * BK);   // async prefetch next tile
    if (kt + 2 < nk) { // gfx1250 global_prefetch_b8: warm L2 for the K+2 tile
      __builtin_prefetch(A + (long)(blockM + arow) * lda + (kt + 2) * BK + acol, 0, 0);
      __builtin_prefetch(Bm + (long)(blockN + brow) * ldb + (kt + 2) * BK + bcol, 0, 0);
    }
    Frag a[2], b[2];
    for (int i = 0; i < 2; ++i) {
      const bf16* p = &As[buf][wm + i * 16 + rA][khalf * 8];
      a[i].q[0] = *(const uint4*)p;          // K = 8*kh .. 8*kh+7
      a[i].q[1] = *(const uint4*)(p + 16);   // K = 16+8*kh .. +7
    }
    for (int j = 0; j < 2; ++j) {
      const bf16* p = &Bs[buf][wn + j * 16 + rA][khalf * 8];
      b[j].q[0] = *(const uint4*)p;
      b[j].q[1] = *(const uint4*)(p + 16);
    }
    for (int i = 0; i < 2; ++i)
      for (int j = 0; j < 2; ++j)
        acc[i][j] = __builtin_amdgcn_wmma_f32_16x16x32_bf16(
            false, a[i].v, false, b[j].v, (short)0, acc[i][j], false, false);
    wait_async0();        // next tile's async DMA landed in LDS
    __syncthreads();      // + everyone done reading current tile
  }

  // Epilogue: C/D layout: lane(0-15)=N, VGPR r -> M=r (+8 for lanes 16-31)
  const int cm0 = blockM + wm + (lane >> 4) * 8;
  const int cn0 = blockN + wn + (lane & 15);
  for (int i = 0; i < 2; ++i)
    for (int j = 0; j < 2; ++j) {
      int col = cn0 + j * 16;
      if (col >= N) continue;
      float bv = bias ? bias[col] : 0.f;
      for (int r = 0; r < 8; ++r) {
        int row = cm0 + i * 16 + r;
        if (row >= M) continue;
        float v = acc[i][j][r] + bv;
        if (RELU) v = v > 0.f ? v : 0.f;
        C[(long)row * ldc + col] = (OutT)v;
      }
    }
}

// ---------------------------------------------------------------------------
// Elementwise / data-movement kernels
// ---------------------------------------------------------------------------
__global__ void k_f2bf(const float* __restrict__ s, bf16* __restrict__ d, long n) {
  long i = (long)blockIdx.x * blockDim.x + threadIdx.x;
  if (i < n) d[i] = (bf16)s[i];
}

__global__ void k_zero(float* p, long n) {
  long i = (long)blockIdx.x * blockDim.x + threadIdx.x;
  if (i < n) p[i] = 0.f;
}

// OIHW (O,C,3,3) -> [O][(ky*3+kx)*C + c] bf16 (matches NHWC im2col K order)
__global__ void k_repack_w(const float* __restrict__ w, bf16* __restrict__ d, int O, int C) {
  long n = (long)O * C * 9;
  long i = (long)blockIdx.x * blockDim.x + threadIdx.x;
  if (i >= n) return;
  int c = (int)(i % C); long t = i / C;
  int k = (int)(t % 9); int o = (int)(t / 9);
  d[i] = (bf16)w[((long)o * C + c) * 9 + k];
}

__global__ void k_nchw2nhwc(const float* __restrict__ s, bf16* __restrict__ d, int B, int C, int HW) {
  long n = (long)B * C * HW;
  long i = (long)blockIdx.x * blockDim.x + threadIdx.x;
  if (i >= n) return;
  int c = (int)(i % C); long t = i / C;
  int p = (int)(t % HW); int b = (int)(t / HW);
  d[i] = (bf16)s[((long)b * C + c) * HW + p];
}

// depthwise 3x3, pad 1, NHWC bf16, fp32 weights/bias
__global__ void k_dw3x3(const bf16* __restrict__ in, const float* __restrict__ w,
                        const float* __restrict__ bias, bf16* __restrict__ out,
                        int B, int H, int W, int C) {
  long n = (long)B * H * W * C;
  long i = (long)blockIdx.x * blockDim.x + threadIdx.x;
  if (i >= n) return;
  int c = (int)(i % C); long t = i / C;
  int x = (int)(t % W); t /= W;
  int y = (int)(t % H);
  int b = (int)(t / H);
  float acc = bias[c];
  for (int ky = 0; ky < 3; ++ky) {
    int yy = y + ky - 1; if (yy < 0 || yy >= H) continue;
    for (int kx = 0; kx < 3; ++kx) {
      int xx = x + kx - 1; if (xx < 0 || xx >= W) continue;
      acc += (float)in[(((long)b * H + yy) * W + xx) * C + c] * w[c * 9 + ky * 3 + kx];
    }
  }
  out[i] = (bf16)acc;
}

__global__ void k_maxpool2(const bf16* __restrict__ in, bf16* __restrict__ out,
                           int B, int H, int W, int C) {
  int OH = H / 2, OW = W / 2;
  long n = (long)B * OH * OW * C;
  long i = (long)blockIdx.x * blockDim.x + threadIdx.x;
  if (i >= n) return;
  int c = (int)(i % C); long t = i / C;
  int ox = (int)(t % OW); t /= OW;
  int oy = (int)(t % OH);
  int b = (int)(t / OH);
  float m = -3.0e38f;
  for (int dy = 0; dy < 2; ++dy)
    for (int dx = 0; dx < 2; ++dx) {
      float v = (float)in[(((long)b * H + 2 * oy + dy) * W + 2 * ox + dx) * C + c];
      m = v > m ? v : m;
    }
  out[i] = (bf16)m;
}

// 3x3 pad-0 im2col, NHWC -> rows B*(H-2)*(W-2), cols 9C ordered (ky,kx,c)
__global__ void k_im2col(const bf16* __restrict__ in, bf16* __restrict__ out,
                         int B, int H, int W, int C) {
  int OH = H - 2, OW = W - 2;
  long n = (long)B * OH * OW * 9 * C;
  long i = (long)blockIdx.x * blockDim.x + threadIdx.x;
  if (i >= n) return;
  int c = (int)(i % C); long t = i / C;
  int k = (int)(t % 9); long r = t / 9;
  int ox = (int)(r % OW); r /= OW;
  int oy = (int)(r % OH);
  int b = (int)(r / OH);
  int ky = k / 3, kx = k % 3;
  out[i] = in[(((long)b * H + oy + ky) * W + ox + kx) * C + c];
}

// mean over HW -> write bf16 into concat buffer column block
__global__ void k_avgpool_cat(const bf16* __restrict__ in, bf16* __restrict__ dst,
                              int B, int HW, int C, int ld, int col0, float scale) {
  long n = (long)B * C;
  long i = (long)blockIdx.x * blockDim.x + threadIdx.x;
  if (i >= n) return;
  int c = (int)(i % C); int b = (int)(i / C);
  float s = 0.f;
  for (int p = 0; p < HW; ++p) s += (float)in[((long)b * HW + p) * C + c];
  dst[(long)b * ld + col0 + c] = (bf16)(s * scale);
}

__global__ void k_cat_f32(const float* __restrict__ src, int ds, bf16* __restrict__ dst,
                          int ld, int col0, long R) {
  long n = R * ds;
  long i = (long)blockIdx.x * blockDim.x + threadIdx.x;
  if (i >= n) return;
  long r = i / ds; int j = (int)(i % ds);
  dst[r * ld + col0 + j] = (bf16)src[i];
}

__global__ void k_cat_bf(const bf16* __restrict__ src, int ds, bf16* __restrict__ dst,
                         int ld, int col0, long R) {
  long n = R * ds;
  long i = (long)blockIdx.x * blockDim.x + threadIdx.x;
  if (i >= n) return;
  long r = i / ds; int j = (int)(i % ds);
  dst[r * ld + col0 + j] = src[i];
}

__global__ void k_cat2(const bf16* __restrict__ a, int da, const bf16* __restrict__ b, int db,
                       bf16* __restrict__ dst, long R) {
  int W = da + db;
  long n = R * W;
  long i = (long)blockIdx.x * blockDim.x + threadIdx.x;
  if (i >= n) return;
  long r = i / W; int j = (int)(i % W);
  dst[i] = (j < da) ? a[r * da + j] : b[r * db + (j - da)];
}

// dst[e] = [ x[row[e]] | x[col[e]] | ea[e] ]
__global__ void k_gather3(const bf16* __restrict__ x, int dx, const bf16* __restrict__ ea, int de,
                          const int* __restrict__ ridx, const int* __restrict__ cidx,
                          bf16* __restrict__ dst, int E) {
  int W = 2 * dx + de;
  long n = (long)E * W;
  long i = (long)blockIdx.x * blockDim.x + threadIdx.x;
  if (i >= n) return;
  long e = i / W; int j = (int)(i % W);
  bf16 v;
  if (j < dx)          v = x[(long)ridx[e] * dx + j];
  else if (j < 2 * dx) v = x[(long)cidx[e] * dx + (j - dx)];
  else                 v = ea[e * de + (j - 2 * dx)];
  dst[i] = v;
}

// dst[e] = [ x[col[e]] | e_feat[e] ]
__global__ void k_gather2(const bf16* __restrict__ x, int dx, const bf16* __restrict__ ef, int de,
                          const int* __restrict__ cidx, bf16* __restrict__ dst, int E) {
  int W = dx + de;
  long n = (long)E * W;
  long i = (long)blockIdx.x * blockDim.x + threadIdx.x;
  if (i >= n) return;
  long e = i / W; int j = (int)(i % W);
  dst[i] = (j < dx) ? x[(long)cidx[e] * dx + j] : ef[e * de + (j - dx)];
}

__global__ void k_scatter_add(const float* __restrict__ m, const int* __restrict__ row,
                              float* __restrict__ agg, int E, int D) {
  long n = (long)E * D;
  long i = (long)blockIdx.x * blockDim.x + threadIdx.x;
  if (i >= n) return;
  long e = i / D; int d = (int)(i % D);
  atomicAdd(&agg[(long)row[e] * D + d], m[i]);
}

__global__ void k_count(const int* __restrict__ row, float* __restrict__ cnt, int E) {
  int i = blockIdx.x * blockDim.x + threadIdx.x;
  if (i < E) atomicAdd(&cnt[row[i]], 1.f);
}

// dst[:, col0:col0+D] = agg / max(cnt,1)   (scatter_mean divisor)
__global__ void k_agg_cat(const float* __restrict__ agg, const float* __restrict__ cnt,
                          bf16* __restrict__ dst, int Nr, int D, int ld, int col0) {
  long n = (long)Nr * D;
  long i = (long)blockIdx.x * blockDim.x + threadIdx.x;
  if (i >= n) return;
  int d = (int)(i % D); int nn = (int)(i / D);
  float cv = cnt[nn]; float inv = 1.f / (cv > 1.f ? cv : 1.f);
  dst[(long)nn * ld + col0 + d] = (bf16)(agg[i] * inv);
}

__global__ void k_logsoftmax(const float* __restrict__ in, float* __restrict__ out, int R, int C) {
  int r = blockIdx.x * blockDim.x + threadIdx.x;
  if (r >= R) return;
  float v[8]; float m = -3.0e38f;
  for (int c = 0; c < C; ++c) { v[c] = in[(long)r * C + c]; m = v[c] > m ? v[c] : m; }
  float s = 0.f;
  for (int c = 0; c < C; ++c) s += expf(v[c] - m);
  float ls = logf(s);
  for (int c = 0; c < C; ++c) out[(long)r * C + c] = v[c] - m - ls;
}

// ---------------------------------------------------------------------------
// Host orchestration
// ---------------------------------------------------------------------------
// params flattened in jax pytree (sorted-key) order:
enum {
  P_EC_B1 = 5, P_EC_B2, P_EC_W1, P_EC_W2,
  P_ECNN_CB, P_ECNN_CW,
  P_ECNN_D1_DWB, P_ECNN_D1_DWW, P_ECNN_D1_PWB, P_ECNN_D1_PWW,
  P_ECNN_D2_DWB, P_ECNN_D2_DWW, P_ECNN_D2_PWB, P_ECNN_D2_PWW,
  P_ECNN_D3_DWB, P_ECNN_D3_DWW, P_ECNN_D3_PWB, P_ECNN_D3_PWW,
  P_EJ_B, P_EJ_W,
  P_L0 = 25,                           // 4 layers x 16 leaves
  P_NC_B1 = 89, P_NC_B2, P_NC_W1, P_NC_W2,
  P_NCNN_CB, P_NCNN_CW,
  P_NCNN_D1_DWB, P_NCNN_D1_DWW, P_NCNN_D1_PWB, P_NCNN_D1_PWW,
  P_NCNN_D2_DWB, P_NCNN_D2_DWW, P_NCNN_D2_PWB, P_NCNN_D2_PWW,
  P_NCNN_D3_DWB, P_NCNN_D3_DWW, P_NCNN_D3_PWB, P_NCNN_D3_PWW,
  P_NJ_B = 107, P_NJ_W = 108
};
enum { L_E_B1 = 0, L_E_B2, L_E_BR, L_E_W1, L_E_W2, L_E_WR,
       L_N1_B1, L_N1_B2, L_N1_W1, L_N1_W2, L_N2_B1, L_N2_B2, L_N2_W1, L_N2_W2,
       L_N_BR, L_N_WR };

extern "C" void kernel_launch(void* const* d_in, const int* in_sizes, int n_in,
                              void* d_out, int out_size, void* d_ws, size_t ws_size,
                              hipStream_t stream) {
  (void)in_sizes; (void)n_in; (void)out_size; (void)ws_size;
  const float* x_in  = (const float*)d_in[0];
  const float* ea_in = (const float*)d_in[1];
  const float* nimg  = (const float*)d_in[2];
  const float* eimg  = (const float*)d_in[3];
  const int*   rowi  = (const int*)d_in[4];
  const int*   coli  = rowi + GE;

  char* wsBase = (char*)d_ws;
  size_t wsOff = 0;
  auto walloc = [&](size_t bytes) -> void* {
    void* p = wsBase + wsOff;
    wsOff = (wsOff + bytes + 255) & ~(size_t)255;
    return p;
  };

#define PARF(i) ((const float*)d_in[(i)])
#define EL(kern, total, ...) \
  kern<<<dim3((unsigned)(((long)(total) + 255) / 256)), dim3(256), 0, stream>>>(__VA_ARGS__)

  auto cvtw = [&](int pi, long n) -> bf16* {
    bf16* dd = (bf16*)walloc((size_t)n * sizeof(bf16));
    EL(k_f2bf, n, PARF(pi), dd, n);
    return dd;
  };

  auto gemm = [&](const bf16* A, int lda, const bf16* Bw, int ldb, const float* bias,
                  void* Cp, int ldc, int M, int Nn, int K, int relu, int f32out) {
    dim3 g((unsigned)cdiv_i(Nn, 64), (unsigned)cdiv_i(M, 128)), blk(256);
    if (f32out) {
      if (relu) k_gemm<1, float><<<g, blk, 0, stream>>>(A, lda, Bw, ldb, bias, (float*)Cp, ldc, M, Nn, K);
      else      k_gemm<0, float><<<g, blk, 0, stream>>>(A, lda, Bw, ldb, bias, (float*)Cp, ldc, M, Nn, K);
    } else {
      if (relu) k_gemm<1, bf16><<<g, blk, 0, stream>>>(A, lda, Bw, ldb, bias, (bf16*)Cp, ldc, M, Nn, K);
      else      k_gemm<0, bf16><<<g, blk, 0, stream>>>(A, lda, Bw, ldb, bias, (bf16*)Cp, ldc, M, Nn, K);
    }
  };

  // ----- workspace layout (chunked CNN keeps peak < ~900 MB) -----
  bf16*  P0      = (bf16*)walloc((size_t)2000 * 256 * 128 * 2);
  bf16*  P1      = (bf16*)walloc((size_t)2000 * 256 * 128 * 2);
  bf16*  COLB    = (bf16*)walloc((size_t)2000 * 36 * 1152 * 2);
  bf16*  nodecat = (bf16*)walloc((size_t)GN * 260 * 2);
  bf16*  edgecat = (bf16*)walloc((size_t)GE * 262 * 2);
  bf16*  xa      = (bf16*)walloc((size_t)GN * 1024 * 2);
  bf16*  xb      = (bf16*)walloc((size_t)GN * 1024 * 2);
  bf16*  eaa     = (bf16*)walloc((size_t)GE * 1024 * 2);
  bf16*  eab     = (bf16*)walloc((size_t)GE * 1024 * 2);
  bf16*  bigcat  = (bf16*)walloc((size_t)GE * 3072 * 2);
  bf16*  ehid    = (bf16*)walloc((size_t)GE * 1024 * 2);
  bf16*  e2b     = (bf16*)walloc((size_t)GE * 1024 * 2);
  float* mbuf    = (float*)walloc((size_t)GE * 1024 * 4);
  float* agg     = (float*)walloc((size_t)GN * 1024 * 4);
  bf16*  n2cat   = (bf16*)walloc((size_t)GN * 2048 * 2);
  bf16*  h2      = (bf16*)walloc((size_t)GN * 1024 * 2);
  bf16*  h3      = (bf16*)walloc((size_t)GN * 1024 * 2);
  bf16*  xcat    = (bf16*)walloc((size_t)GN * 2048 * 2);
  float* cnt     = (float*)walloc((size_t)GN * 4);
  bf16*  tcls    = (bf16*)walloc((size_t)GE * 256 * 2);
  float* clsf    = (float*)walloc((size_t)GE * 4 * 4);

  // ----- weight conversion / repack (fp32 -> bf16, conv OIHW -> GEMM order) -----
  bf16* n_pw1 = cvtw(P_NCNN_D1_PWW, 64 * 3);
  bf16* n_pw2 = cvtw(P_NCNN_D2_PWW, 64 * 64);
  bf16* n_pw3 = cvtw(P_NCNN_D3_PWW, 128 * 64);
  bf16* n_cw  = (bf16*)walloc((size_t)256 * 9 * 128 * 2);
  EL(k_repack_w, 256L * 9 * 128, PARF(P_NCNN_CW), n_cw, 256, 128);
  bf16* e_pw1 = cvtw(P_ECNN_D1_PWW, 128 * 3);
  bf16* e_pw2 = cvtw(P_ECNN_D2_PWW, 128 * 128);
  bf16* e_pw3 = cvtw(P_ECNN_D3_PWW, 256 * 128);
  bf16* e_cw  = (bf16*)walloc((size_t)256 * 9 * 256 * 2);
  EL(k_repack_w, 256L * 9 * 256, PARF(P_ECNN_CW), e_cw, 256, 256);

  // ----- CNN tower (chunked over batch), writes pooled features into cat buf -----
  auto run_cnn = [&](const float* img, int B, bf16* catdst, int catld, int catcol,
                     int C1, int C2, int C3, bool extra_pool,
                     const float* dw1w, const float* dw1b, bf16* pw1w, const float* pw1b,
                     const float* dw2w, const float* dw2b, bf16* pw2w, const float* pw2b,
                     const float* dw3w, const float* dw3b, bf16* pw3w, const float* pw3b,
                     bf16* cwp, const float* cb) {
    for (int b0 = 0; b0 < B; b0 += 2000) {
      int CB = (B - b0 < 2000) ? (B - b0) : 2000;
      long px16 = (long)CB * 256;
      EL(k_nchw2nhwc, px16 * 3, img + (long)b0 * 3 * 256, P0, CB, 3, 256);
      EL(k_dw3x3, px16 * 3, P0, dw1w, dw1b, P1, CB, 16, 16, 3);
      gemm(P1, 3, pw1w, 3, pw1b, P0, C1, (int)px16, C1, 3, 0, 0);
      EL(k_dw3x3, px16 * C1, P0, dw2w, dw2b, P1, CB, 16, 16, C1);
      gemm(P1, C1, pw2w, C1, pw2b, P0, C2, (int)px16, C2, C1, 0, 0);
      EL(k_maxpool2, (long)CB * 64 * C2, P0, P1, CB, 16, 16, C2);          // -> 8x8
      EL(k_dw3x3, (long)CB * 64 * C2, P1, dw3w, dw3b, P0, CB, 8, 8, C2);
      gemm(P0, C2, pw3w, C2, pw3b, P1, C3, CB * 64, C3, C2, 0, 0);
      bf16* cin = P1; int Hc = 8;
      if (extra_pool) { EL(k_maxpool2, (long)CB * 16 * C3, P1, P0, CB, 8, 8, C3); cin = P0; Hc = 4; }
      int OH = Hc - 2, K9 = 9 * C3;
      EL(k_im2col, (long)CB * OH * OH * K9, cin, COLB, CB, Hc, Hc, C3);
      bf16* cout = (cin == P0) ? P1 : P0;
      gemm(COLB, K9, cwp, K9, cb, cout, 256, CB * OH * OH, 256, K9, 0, 0);
      EL(k_avgpool_cat, (long)CB * 256, cout, catdst + (long)b0 * catld,
         CB, OH * OH, 256, catld, catcol, 1.f / (float)(OH * OH));
    }
  };

  EL(k_cat_f32, (long)GN * 4, x_in, 4, nodecat, 260, 0, (long)GN);
  run_cnn(nimg, GN, nodecat, 260, 4, 64, 64, 128, false,
          PARF(P_NCNN_D1_DWW), PARF(P_NCNN_D1_DWB), n_pw1, PARF(P_NCNN_D1_PWB),
          PARF(P_NCNN_D2_DWW), PARF(P_NCNN_D2_DWB), n_pw2, PARF(P_NCNN_D2_PWB),
          PARF(P_NCNN_D3_DWW), PARF(P_NCNN_D3_DWB), n_pw3, PARF(P_NCNN_D3_PWB),
          n_cw, PARF(P_NCNN_CB));
  EL(k_cat_f32, (long)GE * 6, ea_in, 6, edgecat, 262, 0, (long)GE);
  run_cnn(eimg, GE, edgecat, 262, 6, 128, 128, 256, true,
          PARF(P_ECNN_D1_DWW), PARF(P_ECNN_D1_DWB), e_pw1, PARF(P_ECNN_D1_PWB),
          PARF(P_ECNN_D2_DWW), PARF(P_ECNN_D2_DWB), e_pw2, PARF(P_ECNN_D2_PWB),
          PARF(P_ECNN_D3_DWW), PARF(P_ECNN_D3_DWB), e_pw3, PARF(P_ECNN_D3_PWB),
          e_cw, PARF(P_ECNN_CB));

  // ----- join layers -----
  bf16* njw = cvtw(P_NJ_W, 256L * 260);
  bf16* ejw = cvtw(P_EJ_W, 256L * 262);
  gemm(nodecat, 260, njw, 260, PARF(P_NJ_B), xa, 256, GN, 256, 260, 0, 0);
  gemm(edgecat, 262, ejw, 262, PARF(P_EJ_B), eaa, 256, GE, 256, 262, 0, 0);

  // ----- scatter_mean divisor -----
  EL(k_zero, GN, cnt, (long)GN);
  EL(k_count, GE, rowi, cnt, GE);

  // ----- 4 GNN layers -----
  const int dinL[4] = {256, 512, 1024, 512};
  const int douL[4] = {512, 1024, 512, 256};
  for (int l = 0; l < 4; ++l) {
    int base = P_L0 + 16 * l;
    int d = dinL[l], o = douL[l], h = d;
    bf16* w_e1  = cvtw(base + L_E_W1,  (long)h * 3 * d);
    bf16* w_e2  = cvtw(base + L_E_W2,  (long)o * h);
    bf16* w_er  = cvtw(base + L_E_WR,  (long)o * (d + o));
    bf16* w_n11 = cvtw(base + L_N1_W1, (long)h * (d + o));
    bf16* w_n12 = cvtw(base + L_N1_W2, (long)o * h);
    bf16* w_n21 = cvtw(base + L_N2_W1, (long)h * (d + o));
    bf16* w_n22 = cvtw(base + L_N2_W2, (long)o * h);
    bf16* w_nr  = cvtw(base + L_N_WR,  (long)o * (d + o));

    // Edge sub-model: e = Lin([mlp2([x[row],x[col],ea]) , ea])
    EL(k_gather3, (long)GE * 3 * d, xa, d, eaa, d, rowi, coli, bigcat, GE);
    gemm(bigcat, 3 * d, w_e1, 3 * d, PARF(base + L_E_B1), ehid, h, GE, h, 3 * d, 1, 0);
    gemm(ehid, h, w_e2, h, PARF(base + L_E_B2), e2b, o, GE, o, h, 0, 0);
    EL(k_cat2, (long)GE * (o + d), e2b, o, eaa, d, bigcat, (long)GE);
    gemm(bigcat, o + d, w_er, o + d, PARF(base + L_E_BR), eab, o, GE, o, o + d, 0, 0);

    // Node sub-model: m = mlp2([x[col], e]); agg = scatter_mean(m, row)
    EL(k_gather2, (long)GE * (d + o), xa, d, eab, o, coli, bigcat, GE);
    gemm(bigcat, d + o, w_n11, d + o, PARF(base + L_N1_B1), ehid, h, GE, h, d + o, 1, 0);
    gemm(ehid, h, w_n12, h, PARF(base + L_N1_B2), mbuf, o, GE, o, h, 0, 1);
    EL(k_zero, (long)GN * o, agg, (long)GN * o);
    EL(k_scatter_add, (long)GE * o, mbuf, rowi, agg, GE, o);
    EL(k_cat_bf, (long)GN * d, xa, d, n2cat, d + o, 0, (long)GN);
    EL(k_agg_cat, (long)GN * o, agg, cnt, n2cat, GN, o, d + o, d);
    gemm(n2cat, d + o, w_n21, d + o, PARF(base + L_N2_B1), h2, h, GN, h, d + o, 1, 0);
    gemm(h2, h, w_n22, h, PARF(base + L_N2_B2), h3, o, GN, o, h, 0, 0);
    EL(k_cat2, (long)GN * (o + d), h3, o, xa, d, xcat, (long)GN);
    gemm(xcat, o + d, w_nr, o + d, PARF(base + L_N_BR), xb, o, GN, o, o + d, 0, 0);

    bf16* t;
    t = xa; xa = xb; xb = t;
    t = eaa; eaa = eab; eab = t;
  }

  // ----- classifiers + log_softmax -----
  bf16* ncw1 = cvtw(P_NC_W1, 256L * 256);
  bf16* ncw2 = cvtw(P_NC_W2, 2L * 256);
  gemm(xa, 256, ncw1, 256, PARF(P_NC_B1), tcls, 256, GN, 256, 256, 1, 0);
  gemm(tcls, 256, ncw2, 256, PARF(P_NC_B2), clsf, 2, GN, 2, 256, 1, 1);
  EL(k_logsoftmax, GN, clsf, (float*)d_out, GN, 2);

  bf16* ecw1 = cvtw(P_EC_W1, 256L * 256);
  bf16* ecw2 = cvtw(P_EC_W2, 4L * 256);
  gemm(eaa, 256, ecw1, 256, PARF(P_EC_B1), tcls, 256, GE, 256, 256, 1, 0);
  gemm(tcls, 256, ecw2, 256, PARF(P_EC_B2), clsf, 4, GE, 4, 256, 1, 1);
  EL(k_logsoftmax, GE, clsf, (float*)d_out + (long)GN * 2, GE, 4);

#undef EL
#undef PARF
}
